// BaichuanAttention_40286793237094
// MI455X (gfx1250) — compile-verified, matrix operand order checked by reading, not run on hardware
//
#include <hip/hip_runtime.h>

// ---------------------------------------------------------------------------
// Problem constants (match reference)
// ---------------------------------------------------------------------------
constexpr int BATCH = 2;
constexpr int SEQ   = 2048;
constexpr int NHEAD = 32;
constexpr int HDIM  = 128;
constexpr int HID   = NHEAD * HDIM;   // 4096
constexpr int H3    = 3 * HID;        // 12288

// ---------------------------------------------------------------------------
// Vector types
// ---------------------------------------------------------------------------
typedef __attribute__((ext_vector_type(16))) __bf16 bf16x16;
typedef __attribute__((ext_vector_type(8)))  __bf16 bf16x8;
typedef __attribute__((ext_vector_type(4)))  __bf16 bf16x4;
typedef __attribute__((ext_vector_type(8)))  float  f32x8;
typedef __attribute__((ext_vector_type(4)))  unsigned int u32x4;
typedef __attribute__((ext_vector_type(8)))  int i32x8;
typedef __attribute__((ext_vector_type(4)))  int i32x4;

#define WMMA_BF16(a, b, c) \
  __builtin_amdgcn_wmma_f32_16x16x32_bf16(false, (a), false, (b), (short)0, (c), false, false)

static __device__ __forceinline__ f32x8 zero_f32x8() {
  f32x8 z;
#pragma unroll
  for (int i = 0; i < 8; ++i) z[i] = 0.0f;
  return z;
}

// ---------------------------------------------------------------------------
// CDNA5 data movers
// ---------------------------------------------------------------------------
static __device__ __forceinline__ uint32_t lds_off(const void* p) {
  // generic LDS pointer: addr[31:0] is the LDS byte address
  return (uint32_t)(uintptr_t)p;
}

// Per-lane async global->LDS copy (GLOBAL_LOAD_ASYNC_TO_LDS_B128, ASYNCcnt)
static __device__ __forceinline__ void async_ld_b128(uint32_t ldsb, const void* gptr) {
  asm volatile("global_load_async_to_lds_b128 %0, %1, off"
               :: "v"(ldsb), "v"(gptr) : "memory");
}
static __device__ __forceinline__ void wait_async0() {
  asm volatile("s_wait_asynccnt 0x0" ::: "memory");
}

// Tensor Data Mover: one 2D tile per descriptor (TENSORcnt).
// D# bit packing per ISA 8.3/8.4; dims & strides in data_size (=2B) units;
// LDS padding: pad_interval=v -> 2^(v+1) DWORDs between pads, pad_amount=v -> v+1 DWORDs.
static __device__ __forceinline__ void tdm_load_2d_bf16(const void* gaddr, uint32_t lds_addr,
                                                        uint32_t tile_d0, uint32_t tile_d1,
                                                        uint32_t tensor_d0, uint32_t tensor_d1,
                                                        uint32_t stride_d0,
                                                        uint32_t pad_interval, uint32_t pad_amount) {
  const unsigned long long ga = (unsigned long long)(uintptr_t)gaddr;
  u32x4 g0;
  g0[0] = 1u;                                        // count=1, is_restore=0, no gather
  g0[1] = lds_addr;                                  // [63:32]  lds_addr (bytes)
  g0[2] = (uint32_t)ga;                              // [95:64]  global_addr lo
  g0[3] = (uint32_t)(ga >> 32) | (2u << 30);         // [120:96] addr hi, [127:126] type=2
  i32x8 g1;
  g1[0] = (int)((1u << 16)                           // data_size = 1 -> 2 bytes
              | (1u << 20)                           // pad_enable
              | (pad_interval << 22)                 // [24:22]
              | (pad_amount << 25));                 // [31:25]
  g1[1] = (int)((tensor_d0 & 0xFFFFu) << 16);        // [47:32]=0, tensor_dim0 lo16
  g1[2] = (int)((tensor_d0 >> 16) | ((tensor_d1 & 0xFFFFu) << 16));
  g1[3] = (int)((tensor_d1 >> 16) | (tile_d0 << 16));// tensor_dim1 hi16 | tile_dim0
  g1[4] = (int)tile_d1;                              // tile_dim1, tile_dim2=0 (2D)
  g1[5] = (int)stride_d0;                            // tensor_dim0_stride lo32
  g1[6] = 0;                                         // stride hi16 | dim1_stride lo16
  g1[7] = 0;
  const i32x4 gz4 = {0, 0, 0, 0};
  const i32x8 gz8 = {0, 0, 0, 0, 0, 0, 0, 0};
  __builtin_amdgcn_tensor_load_to_lds(g0, g1, gz4, gz4, gz8, 0);
}

// Fragment loader for 16-bit A (row = M) and B (row = N) operands.
// Per ISA 16-bit layout: lane L -> row = L&15, K-halves kb = (L>>4)*8,
// halves 0..7 cover k = kb..kb+7, halves 8..15 cover k = 16+kb..16+kb+7.
// Both ranges are contiguous 16-byte chunks -> two b128 loads per lane.
static __device__ __forceinline__ bf16x16 frag_ld(const __bf16* base, int stride, int koff) {
  const int lane = threadIdx.x & 31;
  const int r    = lane & 15;
  const int kb   = (lane >> 4) << 3;
  const __bf16* p = base + (size_t)r * stride + koff + kb;
  bf16x8 lo = *(const bf16x8*)(p);
  bf16x8 hi = *(const bf16x8*)(p + 16);
  bf16x16 out;
#pragma unroll
  for (int i = 0; i < 8; ++i) { out[i] = lo[i]; out[i + 8] = hi[i]; }
  return out;
}

// ---------------------------------------------------------------------------
// fp32 -> bf16 cast (float4 vectorized)
// ---------------------------------------------------------------------------
__global__ __launch_bounds__(256) void cast_f32_to_bf16(const float* __restrict__ in,
                                                        __bf16* __restrict__ out, int n4) {
  int i = blockIdx.x * 256 + threadIdx.x;
  if (i >= n4) return;
  float4 v = ((const float4*)in)[i];
  bf16x4 o;
  o[0] = (__bf16)v.x; o[1] = (__bf16)v.y; o[2] = (__bf16)v.z; o[3] = (__bf16)v.w;
  ((bf16x4*)out)[i] = o;
}

// ---------------------------------------------------------------------------
// bf16 GEMM, f32 accumulate: C[M,N] = A[M,K] @ B[K,N], all row-major.
// 128x256 tile / WG, BK=32, 8 waves (2x4), 64x64 per wave (4x4 WMMA accums:
// 16 WMMA per 8 fragment loads per K-step -> 1:1 wmma:ds_load_b128).
// A tile staged by TDM (one descriptor per wave, 16-row band, LDS row padding
// via pad fields); B tile staged transposed through VGPRs. Double-buffered.
// ---------------------------------------------------------------------------
constexpr int G_BM = 128, G_BN = 256, G_BK = 32;
constexpr int G_SA = G_BK + 8;  // 40 elems = 80B stride (16B aligned, 20 banks)
constexpr int G_SB = G_BK + 8;

template <typename OT>
__global__ __launch_bounds__(256) void gemm_bf16_wmma(const __bf16* __restrict__ A,
                                                      const __bf16* __restrict__ B,
                                                      OT* __restrict__ C,
                                                      int M, int N, int K,
                                                      int lda, int ldb, int ldc) {
  __shared__ __bf16 lA[2][G_BM * G_SA];  // 20 KB
  __shared__ __bf16 lB[2][G_BN * G_SB];  // 40 KB

  const int tid  = threadIdx.x;
  const int lane = tid & 31;
  const int wave = tid >> 5;
  const int wr   = wave >> 2;   // 0..1 -> 64 rows
  const int wc   = wave & 3;    // 0..3 -> 64 cols
  const int m0   = blockIdx.y * G_BM;
  const int n0   = blockIdx.x * G_BN;

  f32x8 acc[4][4];
#pragma unroll
  for (int mi = 0; mi < 4; ++mi)
#pragma unroll
    for (int ni = 0; ni < 4; ++ni) acc[mi][ni] = zero_f32x8();

  const int nk = K / G_BK;

  // A tile 128x32: every wave issues ONE TDM descriptor for its 16-row band
  // (TDM ignores EXEC, so all-wave issue of disjoint bands is the safe form).
  // 64B rows + 16B LDS pad: pad_interval=3 (16 DW), pad_amount=3 (4 DW).
  auto stageA = [&](int kk, int buf) {
    const int k0 = kk * G_BK;
    tdm_load_2d_bf16(A + (size_t)(m0 + wave * 16) * lda + k0,
                     lds_off(&lA[buf][(wave * 16) * G_SA]),
                     /*tile_d0=*/G_BK, /*tile_d1=*/16,
                     /*tensor_d0=*/(uint32_t)lda, /*tensor_d1=*/(uint32_t)M,
                     /*stride_d0=*/(uint32_t)lda,
                     /*pad_interval=*/3, /*pad_amount=*/3);
  };
  // B tile 32x256, stored transposed lB[n][k] (needs VGPR path)
  auto stageB = [&](int kk, int buf) {
    const int k0 = kk * G_BK;
#pragma unroll
    for (int u = tid; u < 1024; u += 256) {
      int k = u >> 5, nc = (u & 31) << 3;
      const __bf16* gp = B + (size_t)(k0 + k) * ldb + n0 + nc;
      if (k0 + 2 * G_BK < K) __builtin_prefetch(gp + (size_t)(2 * G_BK) * ldb, 0, 1);
      bf16x8 v = *(const bf16x8*)gp;
#pragma unroll
      for (int j = 0; j < 8; ++j) lB[buf][(nc + j) * G_SB + k] = v[j];
    }
  };

  stageA(0, 0);
  stageB(0, 0);
  __builtin_amdgcn_s_wait_tensorcnt(0);
  for (int kk = 0; kk < nk; ++kk) {
    __syncthreads();
    if (kk + 1 < nk) { stageA(kk + 1, (kk + 1) & 1); stageB(kk + 1, (kk + 1) & 1); }
    const int buf = kk & 1;
    bf16x16 af[4];
#pragma unroll
    for (int mi = 0; mi < 4; ++mi)
      af[mi] = frag_ld(&lA[buf][(wr * 64 + mi * 16) * G_SA], G_SA, 0);
#pragma unroll
    for (int ni = 0; ni < 4; ++ni) {
      bf16x16 bfr = frag_ld(&lB[buf][(wc * 64 + ni * 16) * G_SB], G_SB, 0);
#pragma unroll
      for (int mi = 0; mi < 4; ++mi)
        acc[mi][ni] = WMMA_BF16(af[mi], bfr, acc[mi][ni]);
    }
    __builtin_amdgcn_s_wait_tensorcnt(0);  // TDM for next buffer done pre-barrier
  }

  // epilogue: C layout -> element r is (row = r + 8*(lane>=16), col = lane&15)
  const int rsel = (lane >> 4) << 3;
#pragma unroll
  for (int mi = 0; mi < 4; ++mi)
#pragma unroll
    for (int ni = 0; ni < 4; ++ni)
#pragma unroll
      for (int r = 0; r < 8; ++r) {
        int row = m0 + wr * 64 + mi * 16 + r + rsel;
        int col = n0 + wc * 64 + ni * 16 + (lane & 15);
        C[(size_t)row * ldc + col] = (OT)acc[mi][ni][r];
      }
}

// ---------------------------------------------------------------------------
// Flash attention with ALiBi + causal mask.
// One WG per (batch, head, 128 q rows); 8 waves, each owns 16 q rows, so
// online softmax is wave-local. KV consumed in 64-key chunks from the QKV
// buffer (paged scatter+gather over unique offsets == identity).
// K chunk staged with async-LDS DMA; V chunk transposed through VGPRs.
// ---------------------------------------------------------------------------
constexpr int A_SK = HDIM + 8;  // 136 (272B stride, conflict-free)
constexpr int A_SV = 64 + 8;    // 72  (144B stride)
constexpr int A_SP = 64 + 8;    // 72

__global__ __launch_bounds__(256) void attn_alibi_kernel(const __bf16* __restrict__ qkv,
                                                         __bf16* __restrict__ out) {
  __shared__ __bf16 lK[64 * A_SK];     // [key][hd]  row-major  (B-layout for Q@K^T)
  __shared__ __bf16 lV[HDIM * A_SV];   // [hd][key]  transposed (B-layout for P@V)
  __shared__ __bf16 lP[8 * 16 * A_SP]; // wave-private P slabs

  const int tid  = threadIdx.x;
  const int lane = tid & 31;
  const int wave = tid >> 5;

  const int idx = blockIdx.x;
  const int qb = idx & 15;
  const int h  = (idx >> 4) & 31;
  const int b  = idx >> 9;
  const int q0 = qb * 128;

  const float slope = exp2f(-0.25f * (float)(h + 1));  // 2^(-8*(h+1)/32)
  const float scale = 0.08838834764831845f;            // 1/sqrt(128)

  // Q fragments straight from global (A-layout rows == row-major rows)
  const __bf16* qbase = qkv + (size_t)(b * SEQ + q0 + wave * 16) * H3 + (size_t)h * HDIM;
  bf16x16 qf[4];
#pragma unroll
  for (int j = 0; j < 4; ++j) qf[j] = frag_ld(qbase, H3, j * 32);

  f32x8 O[8];
#pragma unroll
  for (int t = 0; t < 8; ++t) O[t] = zero_f32x8();
  float m_i[8], l_i[8];
#pragma unroll
  for (int r = 0; r < 8; ++r) { m_i[r] = -1e30f; l_i[r] = 0.0f; }

  const int qr0 = q0 + wave * 16 + ((lane >> 4) << 3);  // q row of acc elem r

  const __bf16* kbase = qkv + (size_t)(b * SEQ) * H3 + HID + (size_t)h * HDIM;
  const __bf16* vbase = qkv + (size_t)(b * SEQ) * H3 + 2 * HID + (size_t)h * HDIM;

  const int nchunks = q0 / 64 + 2;  // keys [0, q0+128) cover all causal keys
  for (int ch = 0; ch < nchunks; ++ch) {
    const int n0 = ch * 64;
    __syncthreads();
    // stage K chunk [64][128] via async global->LDS DMA
#pragma unroll
    for (int u = tid; u < 1024; u += 256) {
      int key = u >> 4, dc = (u & 15) << 3;
      async_ld_b128(lds_off(&lK[key * A_SK + dc]),
                    kbase + (size_t)(n0 + key) * H3 + dc);
    }
    // stage V chunk transposed lV[hd][key] (VGPR path)
#pragma unroll
    for (int u = tid; u < 1024; u += 256) {
      int key = u >> 4, dc = (u & 15) << 3;
      bf16x8 v = *(const bf16x8*)(vbase + (size_t)(n0 + key) * H3 + dc);
#pragma unroll
      for (int j = 0; j < 8; ++j) lV[(dc + j) * A_SV + key] = v[j];
    }
    wait_async0();
    __syncthreads();

    // scores: 4 tiles of 16 keys, K-dim = HDIM via 4 chained WMMAs
    f32x8 s[4];
#pragma unroll
    for (int c = 0; c < 4; ++c) {
      s[c] = zero_f32x8();
#pragma unroll
      for (int j = 0; j < 4; ++j) {
        bf16x16 kf = frag_ld(&lK[(c * 16) * A_SK], A_SK, j * 32);
        s[c] = WMMA_BF16(qf[j], kf, s[c]);
      }
    }

    // scale + alibi + causal mask; chunk row max
    float mx[8];
#pragma unroll
    for (int r = 0; r < 8; ++r) mx[r] = -3.0e38f;
#pragma unroll
    for (int c = 0; c < 4; ++c) {
      const int key = n0 + c * 16 + (lane & 15);
      const float bias = slope * (float)key;
#pragma unroll
      for (int r = 0; r < 8; ++r) {
        float v = s[c][r] * scale + bias;
        if (key > qr0 + r) v = -1e30f;
        s[c][r] = v;
        mx[r] = fmaxf(mx[r], v);
      }
    }
    // row reductions stay inside 16-lane halves (xor masks 1,2,4,8)
#pragma unroll
    for (int r = 0; r < 8; ++r)
#pragma unroll
      for (int off = 1; off < 16; off <<= 1)
        mx[r] = fmaxf(mx[r], __shfl_xor(mx[r], off));

    float corr[8];
#pragma unroll
    for (int r = 0; r < 8; ++r) {
      float mnew = fmaxf(m_i[r], mx[r]);
      corr[r] = __expf(m_i[r] - mnew);
      m_i[r] = mnew;
    }
#pragma unroll
    for (int c = 0; c < 4; ++c)
#pragma unroll
      for (int r = 0; r < 8; ++r) s[c][r] = __expf(s[c][r] - m_i[r]);

#pragma unroll
    for (int r = 0; r < 8; ++r) {
      float ls = s[0][r] + s[1][r] + s[2][r] + s[3][r];
#pragma unroll
      for (int off = 1; off < 16; off <<= 1) ls += __shfl_xor(ls, off);
      l_i[r] = l_i[r] * corr[r] + ls;
    }
#pragma unroll
    for (int t = 0; t < 8; ++t)
#pragma unroll
      for (int r = 0; r < 8; ++r) O[t][r] *= corr[r];

    // store P to wave-private LDS slab (DS ops in-order within a wave)
    __bf16* pw = &lP[wave * 16 * A_SP];
    const int prow = (lane >> 4) << 3;
#pragma unroll
    for (int c = 0; c < 4; ++c)
#pragma unroll
      for (int r = 0; r < 8; ++r)
        pw[(r + prow) * A_SP + c * 16 + (lane & 15)] = (__bf16)s[c][r];

    // O += P @ V  (K-dim = 64 keys -> 2 x K32 steps, N = 128 hd -> 8 tiles)
#pragma unroll
    for (int ks = 0; ks < 2; ++ks) {
      bf16x16 pf = frag_ld(pw, A_SP, ks * 32);
#pragma unroll
      for (int t = 0; t < 8; ++t) {
        bf16x16 vf = frag_ld(&lV[(t * 16) * A_SV], A_SV, ks * 32);
        O[t] = WMMA_BF16(pf, vf, O[t]);
      }
    }
  }

  // normalize and write bf16 attn output [B*S, H]
#pragma unroll
  for (int r = 0; r < 8; ++r) l_i[r] = 1.0f / l_i[r];
#pragma unroll
  for (int t = 0; t < 8; ++t)
#pragma unroll
    for (int r = 0; r < 8; ++r) {
      int qrow = qr0 + r;
      out[(size_t)(b * SEQ + qrow) * HID + (size_t)h * HDIM + t * 16 + (lane & 15)] =
          (__bf16)(O[t][r] * l_i[r]);
    }
}

// ---------------------------------------------------------------------------
// Host launcher
// ---------------------------------------------------------------------------
extern "C" void kernel_launch(void* const* d_in, const int* in_sizes, int n_in,
                              void* d_out, int out_size, void* d_ws, size_t ws_size,
                              hipStream_t stream) {
  (void)in_sizes; (void)n_in; (void)out_size; (void)ws_size;
  const float* hidden = (const float*)d_in[0];
  const float* wpack  = (const float*)d_in[1];
  const float* oproj  = (const float*)d_in[2];
  // d_in[3]=k_cache, d_in[4]=v_cache, d_in[5]=block_offsets:
  // scatter-then-gather over the same unique offsets is the identity on the
  // freshly projected K/V, so attention reads K/V directly from the QKV buffer.
  float* out = (float*)d_out;

  char* ws = (char*)d_ws;
  __bf16* hB   = (__bf16*)(ws + 0);                      //  33,554,432 B
  __bf16* wB   = (__bf16*)(ws + (size_t)33554432);       // 100,663,296 B
  __bf16* oB   = (__bf16*)(ws + (size_t)134217728);      //  33,554,432 B
  __bf16* qkvB = (__bf16*)(ws + (size_t)167772160);      // 100,663,296 B
  __bf16* aB   = (__bf16*)(ws + (size_t)268435456);      //  33,554,432 B (total ~302 MB)

  // 1) cast fp32 operands to bf16
  {
    int n4h = HID * HID / 4;   // hidden / o_proj
    int n4w = HID * H3 / 4;    // w_pack
    cast_f32_to_bf16<<<(n4h + 255) / 256, 256, 0, stream>>>(hidden, hB, n4h);
    cast_f32_to_bf16<<<(n4w + 255) / 256, 256, 0, stream>>>(wpack, wB, n4w);
    cast_f32_to_bf16<<<(n4h + 255) / 256, 256, 0, stream>>>(oproj, oB, n4h);
  }

  // 2) QKV projection: [4096,4096] @ [4096,12288] -> bf16 [4096,12288]
  gemm_bf16_wmma<__bf16><<<dim3(H3 / G_BN, (BATCH * SEQ) / G_BM), 256, 0, stream>>>(
      hB, wB, qkvB, BATCH * SEQ, H3, HID, HID, H3, H3);

  // 3) flash attention with ALiBi: 2*32*16 = 1024 workgroups
  attn_alibi_kernel<<<BATCH * NHEAD * (SEQ / 128), 256, 0, stream>>>(qkvB, aB);

  // 4) output projection: [4096,4096] @ [4096,4096] -> f32 d_out
  gemm_bf16_wmma<float><<<dim3(HID / G_BN, (BATCH * SEQ) / G_BM), 256, 0, stream>>>(
      aB, oB, out, BATCH * SEQ, HID, HID, HID, HID, HID);
}
